// QuantBnDeformConv2d_70901320123224
// MI455X (gfx1250) — matrix operand order repaired and checked by reading, not compiled, not run on hardware
//
#include <hip/hip_runtime.h>
#include <hip/hip_bf16.h>

typedef __attribute__((ext_vector_type(16))) _Float16 v16h;
typedef __attribute__((ext_vector_type(8)))  float    v8f;

// Problem constants (fixed by the reference)
#define NIMG 8
#define CIN  128
#define HH   64
#define WW   64
#define COUT 128
#define K9   9
#define HWSZ 4096        // 64*64
#define KDIM 1152        // 128*9
#define NSTEP 36         // KDIM / 32  (k outer: s = k*4 + cchunk)

// ---------------------------------------------------------------------------
// Kernel 0a: BN fold + per-tensor symmetric quant scale (single workgroup)
// ---------------------------------------------------------------------------
__global__ __launch_bounds__(256) void prep_scale(
    const float* __restrict__ w, const float* __restrict__ gamma,
    const float* __restrict__ beta, const float* __restrict__ mean,
    const float* __restrict__ var, float* __restrict__ sf,
    float* __restrict__ bias, float* __restrict__ qscale) {
  __shared__ float s_sf[COUT];
  __shared__ float red[256];
  int t = threadIdx.x;
  if (t < COUT) {
    float s = gamma[t] / sqrtf(var[t] + 1e-5f);
    s_sf[t] = s;
    sf[t]   = s;
    bias[t] = (0.0f - mean[t]) * s + beta[t];
  }
  __syncthreads();
  float m = 0.0f;
  for (int i = t; i < COUT * CIN * K9; i += 256) {
    int o = i / (CIN * K9);
    m = fmaxf(m, fabsf(w[i] * s_sf[o]));
  }
  red[t] = m;
  __syncthreads();
  for (int off = 128; off > 0; off >>= 1) {
    if (t < off) red[t] = fmaxf(red[t], red[t + off]);
    __syncthreads();
  }
  if (t == 0) qscale[0] = fmaxf(red[0], 1e-8f) / 127.0f;
}

// ---------------------------------------------------------------------------
// Kernel 0b: quantize + pre-swizzle weights into per-lane WMMA A-tile layout.
// Packed index p = ((s*8 + mt)*32 + lane)*16 + elem   (f16 elements)
//   lane<16 : M = lane,      elems 0..7 -> K 0..7,  elems 8..15 -> K 16..23
//   lane>=16: M = lane-16,   elems 0..7 -> K 8..15, elems 8..15 -> K 24..31
//   step s:  k = s>>2, c = (s&3)*32 + kk   (K-dim order = k*128 + c)
// ---------------------------------------------------------------------------
__global__ __launch_bounds__(256) void pack_weights(
    const float* __restrict__ w, const float* __restrict__ sf,
    const float* __restrict__ qscale, _Float16* __restrict__ wpack) {
  int p = blockIdx.x * 256 + threadIdx.x;   // 576*256 = 147456 exactly
  int elem = p & 15;
  int lane = (p >> 4) & 31;
  int mt   = (p >> 9) & 7;
  int s    = p >> 12;
  int k    = s >> 2;
  int c0   = (s & 3) << 5;
  int m    = lane & 15;
  int kk   = (elem & 7) + ((lane >> 4) << 3) + ((elem >> 3) << 4);
  int o = mt * 16 + m;
  int c = c0 + kk;
  float scl = qscale[0];
  float wv  = w[(o * CIN + c) * K9 + k] * sf[o];
  float q   = fminf(fmaxf(rintf(wv / scl), -128.0f), 127.0f) * scl;
  wpack[p] = (_Float16)q;
}

// ---------------------------------------------------------------------------
// Main kernel: deformable-conv GEMM via v_wmma_f32_16x16x32_f16.
// One workgroup (8 waves) = 16 spatial positions x all 128 Cout.
// B tile (32 K-rows x 16 positions) staged in LDS, shared by all 8 waves.
// ---------------------------------------------------------------------------
__global__ __launch_bounds__(256) void dconv_wmma(
    const float* __restrict__ x, const float* __restrict__ offset,
    const _Float16* __restrict__ wpack, const float* __restrict__ bias,
    float* __restrict__ out) {
  __shared__ int   s_idx[K9 * 16][4];
  __shared__ float s_wt [K9 * 16][4];
  __shared__ __align__(32) _Float16 s_b[2][512];   // [buf][lane*16 + elem]

  int t    = threadIdx.x;
  int lane = t & 31;
  int mt   = t >> 5;                 // wave id 0..7 -> Cout tile
  int tile = blockIdx.x;             // 0..2047
  int img  = tile >> 8;              // 256 spatial tiles per image
  int pos_base = (tile & 255) * 16;  // 16 consecutive positions, same row
  int h    = pos_base >> 6;
  int wb   = pos_base & 63;

  // ---- bilinear sample records: 9 kernel taps x 16 positions ----
  if (t < K9 * 16) {
    int k = t >> 4, n = t & 15;
    int ki = k / 3, kj = k % 3;
    int wc = wb + n;
    const float* offp = offset + ((size_t)img * 18 + 2 * k) * HWSZ + h * WW + wc;
    float py = offp[0]    + (float)(ki + h  - 1);
    float px = offp[HWSZ] + (float)(kj + wc - 1);
    float y0f = floorf(py), x0f = floorf(px);
    float wy1 = py - y0f, wx1 = px - x0f;
    float wyv[2] = {1.0f - wy1, wy1};
    float wxv[2] = {1.0f - wx1, wx1};
    int y0 = (int)y0f, x0 = (int)x0f;
#pragma unroll
    for (int dy = 0; dy < 2; ++dy)
#pragma unroll
      for (int dx = 0; dx < 2; ++dx) {
        int yy = y0 + dy, xx = x0 + dx;
        bool v = (yy >= 0) & (yy < HH) & (xx >= 0) & (xx < WW);
        int yc = min(max(yy, 0), HH - 1);
        int xc = min(max(xx, 0), WW - 1);
        s_idx[t][dy * 2 + dx] = yc * WW + xc;
        s_wt [t][dy * 2 + dx] = v ? wyv[dy] * wxv[dx] : 0.0f;
      }
  }
  __syncthreads();

  // B-staging role of this thread: element pair (kk0, kk0+1) at position n.
  int n      = t & 15;
  int kk0    = (t >> 4) << 1;          // 0,2,...,30
  int blane  = n + ((kk0 & 8) << 1);   // lane holding this (kk, n)
  int belem  = (kk0 & 7) + ((kk0 >> 4) << 3); // element within that lane's v16h

  v8f acc = {};

  for (int s = 0; s < NSTEP; ++s) {
    int k  = s >> 2;
    int c0 = (s & 3) << 5;
    // ---- stage B tile (double-buffered, one barrier per step) ----
    {
      const int*   ip = s_idx[k * 16 + n];
      const float* wp = s_wt [k * 16 + n];
      const float* xb = x + (((size_t)img * CIN + (c0 + kk0)) << 12);
      float v0 = wp[0] * xb[ip[0]] + wp[1] * xb[ip[1]] +
                 wp[2] * xb[ip[2]] + wp[3] * xb[ip[3]];
      const float* xb1 = xb + HWSZ;
      float v1 = wp[0] * xb1[ip[0]] + wp[1] * xb1[ip[1]] +
                 wp[2] * xb1[ip[2]] + wp[3] * xb1[ip[3]];
      union { _Float16 hv[2]; unsigned u; } pk;
      pk.hv[0] = (_Float16)v0;
      pk.hv[1] = (_Float16)v1;
      *(unsigned*)&s_b[s & 1][blane * 16 + belem] = pk.u;
    }
    __syncthreads();
    // ---- A from pre-swizzled global (coalesced 32B/lane), B from LDS ----
    v16h av = *(const v16h*)(wpack + (((size_t)(s * 8 + mt) * 32 + lane) << 4));
    v16h bv = *(const v16h*)(&s_b[s & 1][lane * 16]);
    acc = __builtin_amdgcn_wmma_f32_16x16x32_f16(
        false, av, false, bv, (short)0, acc, false, false);
  }

  // ---- epilogue: D layout -> VGPR r: M = r + 8*(lane>=16), N = lane&15 ----
  int on  = lane & 15;
  int mhi = (lane >> 4) * 8;
  float* outp = out + ((size_t)img * COUT + mt * 16) * HWSZ + pos_base + on;
#pragma unroll
  for (int r = 0; r < 8; ++r) {
    int m = mhi + r;
    outp[(size_t)m * HWSZ] = acc[r] + bias[mt * 16 + m];
  }
}

// ---------------------------------------------------------------------------
extern "C" void kernel_launch(void* const* d_in, const int* in_sizes, int n_in,
                              void* d_out, int out_size, void* d_ws, size_t ws_size,
                              hipStream_t stream) {
  const float* x     = (const float*)d_in[0];
  const float* off   = (const float*)d_in[1];
  const float* cw    = (const float*)d_in[2];
  const float* gamma = (const float*)d_in[3];
  const float* beta  = (const float*)d_in[4];
  const float* mean  = (const float*)d_in[5];
  const float* var   = (const float*)d_in[6];

  char*     ws     = (char*)d_ws;
  float*    sf     = (float*)ws;                 // 128 f32
  float*    bias   = (float*)(ws + 512);         // 128 f32
  float*    qscale = (float*)(ws + 1024);        // 1 f32
  _Float16* wpack  = (_Float16*)(ws + 4096);     // 147456 f16 = 288 KB

  prep_scale  <<<1,    256, 0, stream>>>(cw, gamma, beta, mean, var, sf, bias, qscale);
  pack_weights<<<576,  256, 0, stream>>>(cw, sf, qscale, wpack);
  dconv_wmma  <<<2048, 256, 0, stream>>>(x, off, wpack, bias, (float*)d_out);
}
